// CrossModalAttnMP_64982855188839
// MI455X (gfx1250) — compile-verified, matrix operand order checked by reading, not run on hardware
//
#include <hip/hip_runtime.h>

// ---------------------------------------------------------------------------
// CrossModalAttnMP for MI455X (gfx1250), bf16 WMMA path with f32 accumulate.
// B=64, T=1024, O=128, d=128. All GEMMs tiled 128x128 per 256-thread block
// (8 wave32 waves, each wave owns a 16x128 strip = 8 accumulators).
// bf16 B-operand tiles staged into LDS via TDM (tensor_load_to_lds) or async
// global->LDS copies when available, then swizzled into the
// V_WMMA_F32_16X16X32_BF16 fragment layout (ISA 7.12.2).
// ---------------------------------------------------------------------------

typedef __bf16 bf16_t;
typedef __attribute__((ext_vector_type(16))) __bf16 v16bf;
typedef __attribute__((ext_vector_type(8)))  float  v8f;
typedef unsigned int u32x4 __attribute__((ext_vector_type(4)));
typedef int          i32x4 __attribute__((ext_vector_type(4)));
typedef int          i32x8 __attribute__((ext_vector_type(8)));

typedef __attribute__((address_space(3))) void lds_void_t;
typedef __attribute__((address_space(1))) void gbl_void_t;

#define NT 256  // threads per block (8 waves)

#ifndef __has_builtin
#define __has_builtin(x) 0
#endif

#if __has_builtin(__builtin_amdgcn_tensor_load_to_lds) && \
    __has_builtin(__builtin_amdgcn_s_wait_tensorcnt)
#define HAVE_TDM 1
#endif
#if __has_builtin(__builtin_amdgcn_global_load_async_to_lds_b128) && \
    __has_builtin(__builtin_amdgcn_s_wait_asynccnt)
#define HAVE_ASYNC 1
#endif
#if defined(HAVE_TDM) || defined(HAVE_ASYNC)
#define STAGED 1
#endif

__device__ __forceinline__ v8f wmma_bf16(v16bf a, v16bf b, v8f c) {
  // (neg_a, A, neg_b, B, c_mod, C, reuse_a, reuse_b)
  return __builtin_amdgcn_wmma_f32_16x16x32_bf16(false, a, false, b, (short)0, c,
                                                 false, false);
}

// ---------------------------------------------------------------------------
// Synchronous fill: global (f32 or bf16) -> LDS pre-swizzled B fragments.
//   frag[kt][nt]: lane<16  -> col n = nt*16+lane, elems 0..7 = K kt*32+0..7,
//                                                elems 8..15 = K kt*32+16..23
//                 lane>=16 -> col n = nt*16+lane-16, K +8 and +24 runs.
// ---------------------------------------------------------------------------
template <typename SRC>
__device__ __forceinline__ void fill_bfrag(bf16_t (&Bf)[4][8][32][16],
                                           const SRC* __restrict__ W, int ldw,
                                           int krow0, int tid) {
  for (int e = tid; e < 128 * 128; e += NT) {
    const int n = e & 127;   // column
    const int k = e >> 7;    // row within this 128-row chunk
    const bf16_t v = (bf16_t)(float)W[(size_t)(krow0 + k) * ldw + n];
    const int kt = k >> 5;
    const int kl = k & 31;
    const int i  = (kl & 7) | ((kl & 16) ? 8 : 0);
    const int ln = (n & 15) | ((kl & 8) ? 16 : 0);
    Bf[kt][n >> 4][ln][i] = v;
  }
}

#ifdef STAGED
// ---------------------------------------------------------------------------
// Staged path for bf16 sources: DMA a raw 128x128 bf16 tile (32 KB, tile ==
// tensor, contiguous) into LDS, then swizzle LDS->LDS into fragment layout.
// ---------------------------------------------------------------------------
#ifdef HAVE_TDM
__device__ __forceinline__ void tdm_load_tile_128x128_bf16(const bf16_t* src,
                                                           bf16_t* ldsDst) {
  const unsigned long long ga = (unsigned long long)(const void*)src;
  const unsigned lds = (unsigned)(unsigned long long)(lds_void_t*)ldsDst;
  u32x4 g0;
  g0[0] = 1u;                                    // count=1, user desc, no gather
  g0[1] = lds;                                   // lds_addr
  g0[2] = (unsigned)ga;                          // global_addr[31:0]
  g0[3] = (unsigned)((ga >> 32) & 0x01FFFFFFull) // global_addr[56:32]
        | (2u << 30);                            // type = 2 ("image")
  i32x8 g1;
  g1[0] = (int)(1u << 16);    // workgroup_mask=0, data_size=1 (2 bytes)
  g1[1] = (int)(128u << 16);  // tensor_dim0[15:0]=128 at bits 63:48
  g1[2] = (int)(128u << 16);  // tensor_dim0 hi=0 | tensor_dim1[15:0]=128
  g1[3] = (int)(128u << 16);  // tensor_dim1 hi=0 | tile_dim0=128
  g1[4] = (int)128u;          // tile_dim1=128, tile_dim2=0
  g1[5] = (int)128u;          // tensor_dim0_stride = 128 elements
  g1[6] = 0;                  // stride hi | tensor_dim1_stride lo (unused, 2D)
  g1[7] = 0;
  const i32x4 z4 = {0, 0, 0, 0};
#if __clang_major__ >= 23
  const i32x8 z8 = {0, 0, 0, 0, 0, 0, 0, 0};
  __builtin_amdgcn_tensor_load_to_lds(g0, g1, z4, z4, z8, 0);
#else
  __builtin_amdgcn_tensor_load_to_lds(g0, g1, z4, z4, 0);
#endif
}
#endif  // HAVE_TDM

// Copy 32 KB raw tile into LDS and wait; ends with a full-block barrier.
__device__ __forceinline__ void stage_chunk(bf16_t (&Raw)[128 * 128],
                                            const bf16_t* __restrict__ src,
                                            int tid) {
#ifdef HAVE_TDM
  if (tid < 32) {
    tdm_load_tile_128x128_bf16(src, &Raw[0]);
    __builtin_amdgcn_s_wait_tensorcnt(0);
  }
  __syncthreads();
#else  // HAVE_ASYNC
#pragma unroll
  for (int it = 0; it < 8; ++it) {
    const int e = (it * NT + tid) * 8;  // 8 bf16 = 16 bytes per packet
    __builtin_amdgcn_global_load_async_to_lds_b128(
        (gbl_void_t*)(src + e), (lds_void_t*)(&Raw[e]), 0, 0);
  }
  __builtin_amdgcn_s_wait_asynccnt(0);
  __syncthreads();
#endif
}

__device__ __forceinline__ void swizzle_raw_to_frag(bf16_t (&Bf)[4][8][32][16],
                                                    const bf16_t (&Raw)[128 * 128],
                                                    int tid) {
  for (int e = tid; e < 128 * 128; e += NT) {
    const int n = e & 127;
    const int k = e >> 7;
    const int kt = k >> 5;
    const int kl = k & 31;
    const int i  = (kl & 7) | ((kl & 16) ? 8 : 0);
    const int ln = (n & 15) | ((kl & 8) ? 16 : 0);
    Bf[kt][n >> 4][ln][i] = Raw[e];
  }
}
#endif  // STAGED

// A fragment (16x32, bf16) from row-major f32 source.
__device__ __forceinline__ v16bf a_frag_f32(const float* __restrict__ A, int lda,
                                            int m0, int k0, int lane) {
  const int row = m0 + (lane & 15);
  const int kb  = k0 + ((lane & 16) ? 8 : 0);
  const float* p = A + (size_t)row * lda + kb;
  v16bf a;
#pragma unroll
  for (int i = 0; i < 8; ++i) {
    a[i]     = (bf16_t)p[i];
    a[i + 8] = (bf16_t)p[i + 16];
  }
  return a;
}

// A fragment from row-major bf16 source.
__device__ __forceinline__ v16bf a_frag_bf16(const bf16_t* __restrict__ A, int lda,
                                             int m0, int k0, int lane) {
  const int row = m0 + (lane & 15);
  const int kb  = k0 + ((lane & 16) ? 8 : 0);
  const bf16_t* p = A + (size_t)row * lda + kb;
  v16bf a;
#pragma unroll
  for (int i = 0; i < 8; ++i) {
    a[i]     = p[i];
    a[i + 8] = p[i + 16];
  }
  return a;
}

// A fragment of the TRANSPOSE of a row-major f32 matrix: logical A[m][k] = S[k][m].
__device__ __forceinline__ v16bf a_frag_f32_T(const float* __restrict__ S, int lds_,
                                              int m0, int k0, int lane) {
  const int col = m0 + (lane & 15);
  const int kb  = k0 + ((lane & 16) ? 8 : 0);
  const float* p = S + (size_t)kb * lds_ + col;
  v16bf a;
#pragma unroll
  for (int i = 0; i < 8; ++i) {
    a[i]     = (bf16_t)p[(size_t)i * lds_];
    a[i + 8] = (bf16_t)p[(size_t)(i + 16) * lds_];
  }
  return a;
}

// 8 WMMAs: one A fragment against all 8 N-tiles of the staged B chunk.
__device__ __forceinline__ void mma_rows(v8f (&acc)[8], v16bf a,
                                         const bf16_t (&Bf)[4][8][32][16],
                                         int kt, int lane) {
#pragma unroll
  for (int nt = 0; nt < 8; ++nt) {
    v16bf b = *(const v16bf*)(&Bf[kt][nt][lane][0]);
    acc[nt] = wmma_bf16(a, b, acc[nt]);
  }
}

__device__ __forceinline__ void zero_acc(v8f (&acc)[8]) {
  v8f z = {0.f, 0.f, 0.f, 0.f, 0.f, 0.f, 0.f, 0.f};
#pragma unroll
  for (int nt = 0; nt < 8; ++nt) acc[nt] = z;
}

// ---------------------------------------------------------------------------
// Kernel 1: msg = A(f32, rows x 128) @ W(f32,128x128) + bias  -> bf16 out.
// Used for msg_t (rows=1024, bpb=8) and msg_o (rows=128, bpb=1).
// ---------------------------------------------------------------------------
__global__ __launch_bounds__(NT) void k_msg(const float* __restrict__ A,
                                            const float* __restrict__ W,
                                            const float* __restrict__ bias,
                                            bf16_t* __restrict__ out, int bpb) {
  __shared__ __align__(32) bf16_t Bf[4][8][32][16];
  const int tid = threadIdx.x, lane = tid & 31, wave = tid >> 5;
  const int b = blockIdx.x / bpb, rb = blockIdx.x % bpb;
  const size_t base = (size_t)b * bpb * 128 * 128;
  A += base;
  out += base;
  fill_bfrag(Bf, W, 128, 0, tid);
  __syncthreads();
  const int m0 = rb * 128 + wave * 16;
  v8f acc[8];
  zero_acc(acc);
#pragma unroll
  for (int kt = 0; kt < 4; ++kt) {
    v16bf a = a_frag_f32(A, 128, m0, kt * 32, lane);
    mma_rows(acc, a, Bf, kt, lane);
  }
  const int cb = lane & 15, h8 = (lane & 16) ? 8 : 0;
#pragma unroll
  for (int nt = 0; nt < 8; ++nt) {
    const int col = nt * 16 + cb;
    const float bv = bias[col];
#pragma unroll
    for (int r = 0; r < 8; ++r)
      out[(size_t)(m0 + h8 + r) * 128 + col] = (bf16_t)(acc[nt][r] + bv);
  }
}

// ---------------------------------------------------------------------------
// Kernel 2: agg_to_t = attn(f32, 1024x128) @ msg_o(bf16, 128x128) -> bf16.
// msg_o tile staged via TDM/async copy when available.
// ---------------------------------------------------------------------------
__global__ __launch_bounds__(NT) void k_agg_t(const float* __restrict__ attn,
                                              const bf16_t* __restrict__ msg_o,
                                              bf16_t* __restrict__ out) {
  __shared__ __align__(32) bf16_t Bf[4][8][32][16];
  const int tid = threadIdx.x, lane = tid & 31, wave = tid >> 5;
  const int b = blockIdx.x >> 3, rb = blockIdx.x & 7;
  attn += (size_t)b * 1024 * 128;
  out  += (size_t)b * 1024 * 128;
#ifdef STAGED
  __shared__ __align__(16) bf16_t Raw[128 * 128];
  stage_chunk(Raw, msg_o + (size_t)b * 128 * 128, tid);
  swizzle_raw_to_frag(Bf, Raw, tid);
#else
  fill_bfrag(Bf, msg_o + (size_t)b * 128 * 128, 128, 0, tid);
#endif
  __syncthreads();
  const int m0 = rb * 128 + wave * 16;
  v8f acc[8];
  zero_acc(acc);
#pragma unroll
  for (int kt = 0; kt < 4; ++kt) {
    v16bf a = a_frag_f32(attn, 128, m0, kt * 32, lane);
    mma_rows(acc, a, Bf, kt, lane);
  }
  const int cb = lane & 15, h8 = (lane & 16) ? 8 : 0;
#pragma unroll
  for (int nt = 0; nt < 8; ++nt) {
    const int col = nt * 16 + cb;
#pragma unroll
    for (int r = 0; r < 8; ++r)
      out[(size_t)(m0 + h8 + r) * 128 + col] = (bf16_t)acc[nt][r];
  }
}

// ---------------------------------------------------------------------------
// Kernel 3: agg_to_o = attn^T(128x1024) @ msg_t(bf16, 1024x128) -> bf16.
// K=1024 streamed through LDS in 128-row chunks (TDM/async staged).
// ---------------------------------------------------------------------------
__global__ __launch_bounds__(NT) void k_agg_o(const float* __restrict__ attn,
                                              const bf16_t* __restrict__ msg_t,
                                              bf16_t* __restrict__ out) {
  __shared__ __align__(32) bf16_t Bf[4][8][32][16];
#ifdef STAGED
  __shared__ __align__(16) bf16_t Raw[128 * 128];
#endif
  const int tid = threadIdx.x, lane = tid & 31, wave = tid >> 5;
  const int b = blockIdx.x;
  attn  += (size_t)b * 1024 * 128;
  msg_t += (size_t)b * 1024 * 128;
  out   += (size_t)b * 128 * 128;
  const int m0 = wave * 16;
  v8f acc[8];
  zero_acc(acc);
  for (int kc = 0; kc < 8; ++kc) {
#ifdef STAGED
    stage_chunk(Raw, msg_t + (size_t)kc * 128 * 128, tid);  // ends in barrier
    swizzle_raw_to_frag(Bf, Raw, tid);
#else
    __syncthreads();
    fill_bfrag(Bf, msg_t, 128, kc * 128, tid);
#endif
    __syncthreads();
#pragma unroll
    for (int kt = 0; kt < 4; ++kt) {
      v16bf a = a_frag_f32_T(attn, 128, m0, kc * 128 + kt * 32, lane);
      mma_rows(acc, a, Bf, kt, lane);
    }
#ifndef STAGED
    __syncthreads();
#endif
  }
  const int cb = lane & 15, h8 = (lane & 16) ? 8 : 0;
#pragma unroll
  for (int nt = 0; nt < 8; ++nt) {
    const int col = nt * 16 + cb;
#pragma unroll
    for (int r = 0; r < 8; ++r)
      out[(size_t)(m0 + h8 + r) * 128 + col] = (bf16_t)acc[nt][r];
  }
}

// ---------------------------------------------------------------------------
// Kernel 4: fused update MLP + residual + LayerNorm.
//   y = LN( X + relu( [X | AG] @ W(256x128) + bias ) ) * g + beta
// ---------------------------------------------------------------------------
__global__ __launch_bounds__(NT) void k_update(const float* __restrict__ X,
                                               const bf16_t* __restrict__ AG,
                                               const float* __restrict__ W,
                                               const float* __restrict__ bias,
                                               const float* __restrict__ gam,
                                               const float* __restrict__ bet,
                                               float* __restrict__ out, int bpb) {
  __shared__ __align__(32) bf16_t Bf[4][8][32][16];
  const int tid = threadIdx.x, lane = tid & 31, wave = tid >> 5;
  const int b = blockIdx.x / bpb, rb = blockIdx.x % bpb;
  const size_t base = (size_t)b * bpb * 128 * 128;
  X += base;
  AG += base;
  out += base;
  const int m0 = rb * 128 + wave * 16;
  const int cb = lane & 15, h8 = (lane & 16) ? 8 : 0;
  v8f acc[8];
  zero_acc(acc);

  // phase 0: X against W rows [0,128)
  fill_bfrag(Bf, W, 128, 0, tid);
  __syncthreads();
#pragma unroll
  for (int kt = 0; kt < 4; ++kt) {
    v16bf a = a_frag_f32(X, 128, m0, kt * 32, lane);
    mma_rows(acc, a, Bf, kt, lane);
  }
  // prefetch residual rows for the epilogue (global_prefetch_b8)
#pragma unroll
  for (int r = 0; r < 8; ++r)
    __builtin_prefetch(&X[(size_t)(m0 + h8 + r) * 128 + cb], 0, 1);
  __syncthreads();
  // phase 1: AG against W rows [128,256)
  fill_bfrag(Bf, W, 128, 128, tid);
  __syncthreads();
#pragma unroll
  for (int kt = 0; kt < 4; ++kt) {
    v16bf a = a_frag_bf16(AG, 128, m0, kt * 32, lane);
    mma_rows(acc, a, Bf, kt, lane);
  }

  // epilogue: bias + relu + residual, then per-row LayerNorm
  float gv[8], bv[8], biv[8];
#pragma unroll
  for (int nt = 0; nt < 8; ++nt) {
    const int col = nt * 16 + cb;
    gv[nt]  = gam[col];
    bv[nt]  = bet[col];
    biv[nt] = bias[col];
  }
#pragma unroll
  for (int nt = 0; nt < 8; ++nt) {
    const int col = nt * 16 + cb;
#pragma unroll
    for (int r = 0; r < 8; ++r) {
      const int row = m0 + h8 + r;
      float v = acc[nt][r] + biv[nt];
      v = fmaxf(v, 0.0f);
      v += X[(size_t)row * 128 + col];
      acc[nt][r] = v;  // reuse accumulators as post-residual activations
    }
  }
  // row statistics: reduce over 8 nt-tiles in-register, then across the
  // 16-lane half-wave (xor masks 1,2,4,8 never cross the half boundary).
  float sum[8], ssq[8];
#pragma unroll
  for (int r = 0; r < 8; ++r) {
    float s = 0.f, q = 0.f;
#pragma unroll
    for (int nt = 0; nt < 8; ++nt) {
      const float v = acc[nt][r];
      s += v;
      q += v * v;
    }
    sum[r] = s;
    ssq[r] = q;
  }
#pragma unroll
  for (int m = 1; m < 16; m <<= 1) {
#pragma unroll
    for (int r = 0; r < 8; ++r) {
      sum[r] += __shfl_xor(sum[r], m, 32);
      ssq[r] += __shfl_xor(ssq[r], m, 32);
    }
  }
  float mu[8], rs[8];
#pragma unroll
  for (int r = 0; r < 8; ++r) {
    const float m_ = sum[r] * (1.0f / 128.0f);
    const float v_ = ssq[r] * (1.0f / 128.0f) - m_ * m_;
    mu[r] = m_;
    rs[r] = rsqrtf(v_ + 1e-5f);
  }
#pragma unroll
  for (int nt = 0; nt < 8; ++nt) {
    const int col = nt * 16 + cb;
#pragma unroll
    for (int r = 0; r < 8; ++r) {
      const int row = m0 + h8 + r;
      out[(size_t)row * 128 + col] = (acc[nt][r] - mu[r]) * rs[r] * gv[nt] + bv[nt];
    }
  }
}

// ---------------------------------------------------------------------------
// Host-side launch
// ---------------------------------------------------------------------------
extern "C" void kernel_launch(void* const* d_in, const int* in_sizes, int n_in,
                              void* d_out, int out_size, void* d_ws, size_t ws_size,
                              hipStream_t stream) {
  (void)in_sizes; (void)n_in; (void)out_size; (void)ws_size;

  const float* token   = (const float*)d_in[0];   // [64,1024,128]
  const float* obj     = (const float*)d_in[1];   // [64,128,128]
  const float* attn    = (const float*)d_in[2];   // [64,1024,128]
  const float* W_msg_o = (const float*)d_in[3];
  const float* b_msg_o = (const float*)d_in[4];
  const float* W_msg_t = (const float*)d_in[5];
  const float* b_msg_t = (const float*)d_in[6];
  const float* W_upd_t = (const float*)d_in[7];   // [256,128]
  const float* b_upd_t = (const float*)d_in[8];
  const float* W_upd_o = (const float*)d_in[9];   // [256,128]
  const float* b_upd_o = (const float*)d_in[10];
  const float* g_t     = (const float*)d_in[11];
  const float* be_t    = (const float*)d_in[12];
  const float* g_o     = (const float*)d_in[13];
  const float* be_o    = (const float*)d_in[14];

  float* out_t = (float*)d_out;                       // [64,1024,128]
  float* out_o = out_t + (size_t)64 * 1024 * 128;     // [64,128,128]

  // workspace: bf16 intermediates (~38 MB total)
  bf16_t* w_msg_t = (bf16_t*)d_ws;                                 // 64*1024*128
  bf16_t* w_msg_o = w_msg_t + (size_t)64 * 1024 * 128;             // 64*128*128
  bf16_t* w_agg_t = w_msg_o + (size_t)64 * 128 * 128;              // 64*1024*128
  bf16_t* w_agg_o = w_agg_t + (size_t)64 * 1024 * 128;             // 64*128*128

  const dim3 blk(NT);
  // messages
  k_msg<<<512, blk, 0, stream>>>(token, W_msg_t, b_msg_t, w_msg_t, 8);
  k_msg<<<64,  blk, 0, stream>>>(obj,   W_msg_o, b_msg_o, w_msg_o, 1);
  // aggregation
  k_agg_t<<<512, blk, 0, stream>>>(attn, w_msg_o, w_agg_t);
  k_agg_o<<<64,  blk, 0, stream>>>(attn, w_msg_t, w_agg_o);
  // fused update + residual + LayerNorm
  k_update<<<512, blk, 0, stream>>>(token, w_agg_t, W_upd_t, b_upd_t, g_t, be_t, out_t, 8);
  k_update<<<64,  blk, 0, stream>>>(obj,   w_agg_o, W_upd_o, b_upd_o, g_o, be_o, out_o, 1);
}
